// SPF_33105607918153
// MI455X (gfx1250) — compile-verified, matrix operand order checked by reading, not run on hardware
//
#include <hip/hip_runtime.h>

typedef unsigned int   u32;
typedef unsigned short u16;
typedef __attribute__((ext_vector_type(16))) __bf16 v16bf;
typedef __attribute__((ext_vector_type(8)))  float  v8f;
typedef __attribute__((ext_vector_type(4)))  u32    v4u;
typedef __attribute__((ext_vector_type(2)))  u32    v2u;
typedef __attribute__((ext_vector_type(4)))  float  v4f;

// exact pointee type the async-LDS builtin wants: gcc-vector int4
typedef int v4i_g __attribute__((vector_size(16)));
typedef __attribute__((address_space(1))) v4i_g* gv4i_p;  // global int4*
typedef __attribute__((address_space(3))) v4i_g* lv4i_p;  // LDS int4*

#define SL 2048
#define BS 2
#define E  1024
#define H  16
#define PF 16
#define DH 64
#define T  (SL*BS)   // 4096 tokens

// ---- CDNA5 async global->LDS path (guarded; falls back to vector copies) ----
#if defined(__gfx1250__) && __has_builtin(__builtin_amdgcn_global_load_async_to_lds_b128)
#define HAVE_ASYNC_LDS 1
#define ASYNC_B128(lds, g) \
  __builtin_amdgcn_global_load_async_to_lds_b128((gv4i_p)(g), (lv4i_p)(lds), 0, 0)
#else
#define HAVE_ASYNC_LDS 0
#endif

__device__ __forceinline__ void wait_async0(){
#if defined(__gfx1250__)
#if __has_builtin(__builtin_amdgcn_s_wait_asynccnt)
  __builtin_amdgcn_s_wait_asynccnt(0);
#else
  asm volatile("s_wait_asynccnt 0x0" ::: "memory");
#endif
#endif
}

// ---------- bf16 helpers (manual, RNE) ----------
__device__ __forceinline__ u16 f2bf(float f){
  u32 x = __float_as_uint(f);
  return (u16)((x + 0x7FFFu + ((x >> 16) & 1u)) >> 16);
}
__device__ __forceinline__ float bflo(u32 u){ return __uint_as_float(u << 16); }
__device__ __forceinline__ float bfhi(u32 u){ return __uint_as_float(u & 0xFFFF0000u); }
__device__ __forceinline__ u32 pack2(float a, float b){
  return (u32)f2bf(a) | ((u32)f2bf(b) << 16);
}

// ---------- weight fp32 -> bf16 convert + TRANSPOSE (done once, off hot path) ----
// W[k][n] fp32  ->  Wt[n][k] bf16, LDS-tiled so both global accesses are coalesced.
// Grid: 4 weights x 32x32 tiles of 32x32 = 4096 blocks, 256 threads.
__global__ __launch_bounds__(256) void cvtw_t_kernel(const float* __restrict__ a,
    const float* __restrict__ b, const float* __restrict__ c,
    const float* __restrict__ d, u16* __restrict__ out){
  __shared__ u16 tile[32][33];
  const int bid = blockIdx.x;
  const int w  = bid >> 10;          // which weight
  const int t  = bid & 1023;
  const int tk = t >> 5;             // k-tile
  const int tn = t & 31;             // n-tile
  const float* src = (w == 0) ? a : (w == 1) ? b : (w == 2) ? c : d;
  const int tid = threadIdx.x;
  const int r  = tid >> 3;           // 0..31
  const int c4 = (tid & 7) * 4;      // 0,4,..,28
  v4f v = *(const v4f*)(src + (size_t)(tk*32 + r) * E + tn*32 + c4);
  tile[r][c4+0] = f2bf(v.x); tile[r][c4+1] = f2bf(v.y);
  tile[r][c4+2] = f2bf(v.z); tile[r][c4+3] = f2bf(v.w);
  __syncthreads();
  const u16 t0 = tile[c4+0][r], t1 = tile[c4+1][r];
  const u16 t2 = tile[c4+2][r], t3 = tile[c4+3][r];
  v2u o; o.x = (u32)t0 | ((u32)t1 << 16); o.y = (u32)t2 | ((u32)t3 << 16);
  *(v2u*)(out + (size_t)w * E * E + (size_t)(tn*32 + r) * E + tk*32 + c4) = o;
}

// ---------- LayerNorm: one block per token, bf16 output ----------
__global__ __launch_bounds__(256) void ln_kernel(const float* __restrict__ x,
                                                 const float* __restrict__ g,
                                                 const float* __restrict__ bt,
                                                 u16* __restrict__ xn){
  const int token = blockIdx.x;
  const int tid = threadIdx.x;
  v4f v = ((const v4f*)(x + (size_t)token * E))[tid];
  float s  = v.x + v.y + v.z + v.w;
  float ss = v.x*v.x + v.y*v.y + v.z*v.z + v.w*v.w;
  __shared__ float r1[256];
  __shared__ float r2[256];
  r1[tid] = s; r2[tid] = ss;
  __syncthreads();
  for (int st = 128; st > 0; st >>= 1){
    if (tid < st){ r1[tid] += r1[tid+st]; r2[tid] += r2[tid+st]; }
    __syncthreads();
  }
  const float mean = r1[0] * (1.0f / E);
  const float var  = r2[0] * (1.0f / E) - mean * mean;
  const float rstd = rsqrtf(var + 1e-5f);
  v4f gg = ((const v4f*)g)[tid];
  v4f bb = ((const v4f*)bt)[tid];
  const float y0 = (v.x - mean) * rstd * gg.x + bb.x;
  const float y1 = (v.y - mean) * rstd * gg.y + bb.y;
  const float y2 = (v.z - mean) * rstd * gg.z + bb.z;
  const float y3 = (v.w - mean) * rstd * gg.w + bb.w;
  v2u o; o.x = pack2(y0, y1); o.y = pack2(y2, y3);
  ((v2u*)(xn + (size_t)token * E))[tid] = o;
}

// ---------- bf16 GEMM via v_wmma_f32_16x16x32_bf16 ----------
// C[M,N] = A[M,K] * Bt[N,K]^T + bias (+ residual).
// 128x128 block tile, BK=32, double-buffered LDS, async global->LDS staging.
// 8 waves: wave (wm,wn) owns a 32x64 sub-tile = 2x4 WMMA accumulators.
union Frag { v16bf v; v4u q[2]; };

__global__ __launch_bounds__(256) void gemm_bf16_kernel(
    const u16* __restrict__ A, const u16* __restrict__ Bt,
    const float* __restrict__ bias, const float* __restrict__ residual,
    float* __restrict__ outF, u16* __restrict__ outB,
    const int M, const int N, const int K)
{
  __shared__ __align__(16) u16 sA[2][128][32];  // A tile  [m][k]
  __shared__ __align__(16) u16 sB[2][128][32];  // Bt tile [n][k]
  const int tid  = threadIdx.x;
  const int wave = tid >> 5, lane = tid & 31;
  const int l16  = lane & 15, kh = lane >> 4;   // kh: which 16-lane half
  const int wm   = wave & 3,  wn = wave >> 2;
  const int m0 = blockIdx.y * 128;
  const int n0 = blockIdx.x * 128;

  v8f acc[2][4];
  #pragma unroll
  for (int nt = 0; nt < 4; ++nt){
    const float bv = bias[n0 + wn*64 + nt*16 + l16];
    #pragma unroll
    for (int mt = 0; mt < 2; ++mt)
      #pragma unroll
      for (int r = 0; r < 8; ++r) acc[mt][nt][r] = bv;
  }

  // staging: thread covers 16 contiguous u16 (two b128) of one row in each tile
  const int row = tid >> 1, half = tid & 1;
  const u16* aptr = A  + (size_t)(m0 + row) * K + half * 16;
  const u16* bptr = Bt + (size_t)(n0 + row) * K + half * 16;

  auto stage = [&](int buf, int kk){
#if HAVE_ASYNC_LDS
    ASYNC_B128(&sA[buf][row][half*16],     aptr + kk);
    ASYNC_B128(&sA[buf][row][half*16 + 8], aptr + kk + 8);
    ASYNC_B128(&sB[buf][row][half*16],     bptr + kk);
    ASYNC_B128(&sB[buf][row][half*16 + 8], bptr + kk + 8);
#else
    v4u* ad = (v4u*)&sA[buf][row][half*16];
    const v4u* as = (const v4u*)(aptr + kk);
    ad[0] = as[0]; ad[1] = as[1];
    v4u* bd = (v4u*)&sB[buf][row][half*16];
    const v4u* bs = (const v4u*)(bptr + kk);
    bd[0] = bs[0]; bd[1] = bs[1];
#endif
  };

  stage(0, 0);
#if HAVE_ASYNC_LDS
  wait_async0();
#endif
  __syncthreads();

  int buf = 0;
  for (int kk = 0; kk < K; kk += 32){
    if (kk + 32 < K) stage(buf ^ 1, kk + 32);   // prefetch next tile (async)

    // fragments per ISA 16-bit 16x32 layout:
    // lanes 0-15 (kh=0): K 0..7 then 16..23 ; lanes 16-31 (kh=1): K 8..15 then 24..31
    Frag a[2], bfr[4];
    #pragma unroll
    for (int mt = 0; mt < 2; ++mt){
      const u16* p = &sA[buf][wm*32 + mt*16 + l16][kh * 8];
      a[mt].q[0] = *(const v4u*)p;
      a[mt].q[1] = *(const v4u*)(p + 16);
    }
    #pragma unroll
    for (int nt = 0; nt < 4; ++nt){
      const u16* p = &sB[buf][wn*64 + nt*16 + l16][kh * 8];
      bfr[nt].q[0] = *(const v4u*)p;
      bfr[nt].q[1] = *(const v4u*)(p + 16);
    }
    #pragma unroll
    for (int mt = 0; mt < 2; ++mt)
      #pragma unroll
      for (int nt = 0; nt < 4; ++nt)
        acc[mt][nt] = __builtin_amdgcn_wmma_f32_16x16x32_bf16(
            false, a[mt].v, false, bfr[nt].v, (short)0, acc[mt][nt], false, false);

#if HAVE_ASYNC_LDS
    wait_async0();          // next tile resident before anyone reads it
#endif
    __syncthreads();
    buf ^= 1;
  }

  // epilogue: C/D layout lane l16 = N, kh selects M+8, VGPR r = M row
  #pragma unroll
  for (int mt = 0; mt < 2; ++mt)
    #pragma unroll
    for (int nt = 0; nt < 4; ++nt){
      const int n = n0 + wn*64 + nt*16 + l16;
      #pragma unroll
      for (int r = 0; r < 8; ++r){
        const int m = m0 + wm*32 + mt*16 + kh*8 + r;
        const size_t off = (size_t)m * N + n;
        float val = acc[mt][nt][r];
        if (residual) val += residual[off];
        if (outF) outF[off] = val;
        if (outB) outB[off] = f2bf(val);
      }
    }
}

// ---------- sliding-window attention: one thread per (s,b,h) ----------
__global__ __launch_bounds__(256) void attn_kernel(const u16* __restrict__ Q,
    const u16* __restrict__ Kb, const u16* __restrict__ Vb, u16* __restrict__ AO)
{
  const int idx = blockIdx.x * 256 + threadIdx.x;  // < T*H = 65536
  const int h = idx & (H - 1);
  const int b = (idx >> 4) & (BS - 1);
  const int s = idx >> 5;
  const size_t qoff = (size_t)(s * BS + b) * E + h * DH;

  float q[DH];
  #pragma unroll
  for (int j = 0; j < 8; ++j){
    v4u u = ((const v4u*)(Q + qoff))[j];
    #pragma unroll
    for (int c = 0; c < 4; ++c){
      q[j*8 + 2*c]     = bflo(u[c]);
      q[j*8 + 2*c + 1] = bfhi(u[c]);
    }
  }
  float sc[PF];
  float mx = -3.0e38f;
  #pragma unroll
  for (int p = 0; p < PF; ++p){
    int t = s - (PF - 1) + p; if (t < 0) t += SL;   // (s-15+p) mod SL
    const v4u* kp = (const v4u*)(Kb + (size_t)(t * BS + b) * E + h * DH);
    float d = 0.0f;
    #pragma unroll
    for (int j = 0; j < 8; ++j){
      v4u u = kp[j];
      #pragma unroll
      for (int c = 0; c < 4; ++c){
        d += q[j*8 + 2*c]     * bflo(u[c]);
        d += q[j*8 + 2*c + 1] * bfhi(u[c]);
      }
    }
    sc[p] = d * 0.125f;   // 1/sqrt(DH)
    mx = fmaxf(mx, sc[p]);
  }
  float sum = 0.0f;
  #pragma unroll
  for (int p = 0; p < PF; ++p){ sc[p] = __expf(sc[p] - mx); sum += sc[p]; }
  const float inv = 1.0f / sum;

  float out[DH];
  #pragma unroll
  for (int j = 0; j < DH; ++j) out[j] = 0.0f;
  #pragma unroll
  for (int p = 0; p < PF; ++p){
    int t = s - (PF - 1) + p; if (t < 0) t += SL;
    const v4u* vp = (const v4u*)(Vb + (size_t)(t * BS + b) * E + h * DH);
    const float w = sc[p] * inv;
    #pragma unroll
    for (int j = 0; j < 8; ++j){
      v4u u = vp[j];
      #pragma unroll
      for (int c = 0; c < 4; ++c){
        out[j*8 + 2*c]     += w * bflo(u[c]);
        out[j*8 + 2*c + 1] += w * bfhi(u[c]);
      }
    }
  }
  v4u* op = (v4u*)(AO + qoff);
  #pragma unroll
  for (int j = 0; j < 8; ++j){
    v4u u;
    #pragma unroll
    for (int c = 0; c < 4; ++c)
      u[c] = pack2(out[j*8 + 2*c], out[j*8 + 2*c + 1]);
    op[j] = u;
  }
}

extern "C" void kernel_launch(void* const* d_in, const int* in_sizes, int n_in,
                              void* d_out, int out_size, void* d_ws, size_t ws_size,
                              hipStream_t stream){
  (void)in_sizes; (void)n_in; (void)out_size; (void)ws_size;
  const float* x    = (const float*)d_in[0];
  const float* ln_g = (const float*)d_in[1];
  const float* ln_b = (const float*)d_in[2];
  const float* Wq = (const float*)d_in[3]; const float* bq = (const float*)d_in[4];
  const float* Wk = (const float*)d_in[5]; const float* bk = (const float*)d_in[6];
  const float* Wv = (const float*)d_in[7]; const float* bv = (const float*)d_in[8];
  const float* Wo = (const float*)d_in[9]; const float* bo = (const float*)d_in[10];
  float* out = (float*)d_out;

  const size_t TE = (size_t)T * E;  // 4M elems
  const size_t WE = (size_t)E * E;  // 1M elems
  u16* ws  = (u16*)d_ws;            // total bf16 staging: 24M elems = 48 MB
  u16* xn  = ws;
  u16* Wqt = ws + TE;               // weights stored TRANSPOSED [n][k]
  u16* Wkt = Wqt + WE;
  u16* Wvt = Wkt + WE;
  u16* Wot = Wvt + WE;
  u16* Qb  = ws + TE + 4 * WE;
  u16* Kbb = Qb  + TE;
  u16* Vbb = Kbb + TE;
  u16* AOb = Vbb + TE;

  cvtw_t_kernel<<<4096, 256, 0, stream>>>(Wq, Wk, Wv, Wo, Wqt);
  ln_kernel<<<T, 256, 0, stream>>>(x, ln_g, ln_b, xn);

  dim3 gg(E / 128, T / 128);  // (8, 32) blocks, 8 waves each
  gemm_bf16_kernel<<<gg, 256, 0, stream>>>(xn, Wqt, bq, nullptr, nullptr, Qb,  T, E, E);
  gemm_bf16_kernel<<<gg, 256, 0, stream>>>(xn, Wkt, bk, nullptr, nullptr, Kbb, T, E, E);
  gemm_bf16_kernel<<<gg, 256, 0, stream>>>(xn, Wvt, bv, nullptr, nullptr, Vbb, T, E, E);

  attn_kernel<<<(T * H) / 256, 256, 0, stream>>>(Qb, Kbb, Vbb, AOb);

  gemm_bf16_kernel<<<gg, 256, 0, stream>>>(AOb, Wot, bo, x, out, nullptr, T, E, E);
}
